// PrimeKGDrugRepurposingGNN_12120397709962
// MI455X (gfx1250) — matrix operand order repaired and checked by reading, not compiled, not run on hardware
//
#include <hip/hip_runtime.h>
#include <hip/hip_bf16.h>

#define N_NODES 100000
#define N_TYPES 10
#define N_EDGES 3200000
#define HIDDEN  256
#define EMBED   128
#define N_PAIRS 100000
#define MR      5            // row tiles per wave: 6250 = 5 * 1250 (exact)

typedef __bf16 v16bf __attribute__((ext_vector_type(16)));
typedef float  v8f   __attribute__((ext_vector_type(8)));

union Frag {               // 32 bytes: one WMMA bf16 A/B fragment per lane
  v16bf v;
  uint4 q[2];
};

union BF4 {                // 8 bytes: 4 packed bf16
  ushort4 s;
  __bf16  h[4];
};

// ---------------------------------------------------------------------------
// 1. x = node_emb + type_emb[type_ids]
// ---------------------------------------------------------------------------
__global__ void embed_kernel(const float* __restrict__ node_emb,
                             const float* __restrict__ type_emb,
                             const int* __restrict__ type_ids,
                             float* __restrict__ x) {
  int i = blockIdx.x * blockDim.x + threadIdx.x;
  if (i < N_NODES * HIDDEN) {
    int n = i >> 8;
    int c = i & (HIDDEN - 1);
    x[i] = node_emb[i] + type_emb[type_ids[n] * HIDDEN + c];
  }
}

// ---------------------------------------------------------------------------
// 2. rowptr[n] = lower_bound(rows, n)  (rows is sorted)
// ---------------------------------------------------------------------------
__global__ void rowptr_kernel(const int* __restrict__ rows,
                              int* __restrict__ rowptr) {
  int n = blockIdx.x * blockDim.x + threadIdx.x;
  if (n <= N_NODES) {
    int lo = 0, hi = N_EDGES;
    while (lo < hi) {
      int mid = (lo + hi) >> 1;
      if (rows[mid] < n) lo = mid + 1; else hi = mid;
    }
    rowptr[n] = lo;
  }
}

// ---------------------------------------------------------------------------
// 3. y[row,:] = sum_e val[e] * x[cols[e],:]
//    64 threads per row, float4 per thread (b128 gathers, L2-resident x),
//    bf16 output packed 4-wide (b64 stores) -> feeds WMMA GEMM directly.
// ---------------------------------------------------------------------------
__global__ void spmm_kernel(const float* __restrict__ x,
                            const int* __restrict__ rowptr,
                            const int* __restrict__ cols,
                            const float* __restrict__ vals,
                            __bf16* __restrict__ y) {
  int row = blockIdx.x;
  int c4 = threadIdx.x;                       // 64 threads, 4 floats each
  int s = rowptr[row], e = rowptr[row + 1];
  float4 acc = make_float4(0.f, 0.f, 0.f, 0.f);
  for (int k = s; k < e; ++k) {
    float w = vals[k];
    const float4* xp = (const float4*)(x + (size_t)cols[k] * HIDDEN) + c4;
    float4 v = *xp;
    acc.x = fmaf(w, v.x, acc.x);
    acc.y = fmaf(w, v.y, acc.y);
    acc.z = fmaf(w, v.z, acc.z);
    acc.w = fmaf(w, v.w, acc.w);
  }
  BF4 o;
  o.h[0] = (__bf16)acc.x; o.h[1] = (__bf16)acc.y;
  o.h[2] = (__bf16)acc.z; o.h[3] = (__bf16)acc.w;
  *((ushort4*)(y + (size_t)row * HIDDEN) + c4) = o.s;
}

// ---------------------------------------------------------------------------
// Pre-swizzle fp32 weights [K,N] into per-lane WMMA B-fragment order (bf16):
// each lane's 16-element fragment for one (ktile, coltile) becomes 32
// contiguous bytes -> 2 x global_load_b128 in the GEMM hot loop.
// ---------------------------------------------------------------------------
__global__ void swizzle_w_kernel(const float* __restrict__ W,
                                 __bf16* __restrict__ out, int K, int N) {
  int i = blockIdx.x * blockDim.x + threadIdx.x;
  if (i >= K * N) return;
  int j    = i & 15;
  int lane = (i >> 4) & 31;
  int tile = i >> 9;
  int ntiles = N >> 4;
  int kt = tile / ntiles, ct = tile % ntiles;
  int k   = kt * 32 + (j >> 3) * 16 + (lane >> 4) * 8 + (j & 7);
  int col = ct * 16 + (lane & 15);
  out[i] = (__bf16)W[(size_t)k * N + col];
}

// ---------------------------------------------------------------------------
// 4/6. C = [relu](A @ W + bias)
//      A:[M,K] bf16 row-major, Wswz: pre-swizzled B fragments, C:[M,N] f32.
//      One wave computes an 80x32 macro-tile (5 row-tiles x 2 col-tiles);
//      tiling divides M and N exactly -> no guards, EXEC all-1s throughout.
//      Per K-step: 4 b128 B loads + 10 b128 A loads feed 10 WMMAs.
// ---------------------------------------------------------------------------
__global__ void gemm_bias_wmma(const __bf16* __restrict__ A,
                               const __bf16* __restrict__ Wswz,
                               const float* __restrict__ bias,
                               float* __restrict__ C,
                               int M, int N, int K, int relu) {
  int lane = threadIdx.x & 31;
  int wave = __builtin_amdgcn_readfirstlane(threadIdx.x >> 5);  // scalar
  int gw = blockIdx.x * 8 + wave;            // blockDim.x == 256
  int ncs = N >> 5;                          // 32-col supertiles
  int rs = gw / ncs, cs = gw % ncs;          // all-scalar tile math
  int half = lane >> 4, l15 = lane & 15;
  int ntiles = N >> 4;
  int tc0 = cs * 2, tc1 = cs * 2 + 1;

  v8f acc[MR][2] = {};
  for (int kk = 0; kk < K; kk += 32) {
    int kt = kk >> 5;
    Frag b0, b1;
    const uint4* bp0 = (const uint4*)(Wswz + (((size_t)kt * ntiles + tc0) * 32 + lane) * 16);
    const uint4* bp1 = (const uint4*)(Wswz + (((size_t)kt * ntiles + tc1) * 32 + lane) * 16);
    b0.q[0] = bp0[0]; b0.q[1] = bp0[1];
    b1.q[0] = bp1[0]; b1.q[1] = bp1[1];
#pragma unroll
    for (int r = 0; r < MR; ++r) {
      int arow = (rs * MR + r) * 16 + l15;
      const uint4* ap = (const uint4*)(A + (size_t)arow * K + kk + half * 8);
      Frag a;
      a.q[0] = ap[0];                        // K ka..ka+7
      a.q[1] = ap[2];                        // K ka+16..ka+23
      acc[r][0] = __builtin_amdgcn_wmma_f32_16x16x32_bf16(
          false, a.v, false, b0.v, (short)0, acc[r][0], false, false);
      acc[r][1] = __builtin_amdgcn_wmma_f32_16x16x32_bf16(
          false, a.v, false, b1.v, (short)0, acc[r][1], false, false);
    }
  }

  float bv[2] = { bias[tc0 * 16 + l15], bias[tc1 * 16 + l15] };
#pragma unroll
  for (int r = 0; r < MR; ++r) {
#pragma unroll
    for (int q = 0; q < 2; ++q) {
      int col = (q ? tc1 : tc0) * 16 + l15;
#pragma unroll
      for (int v = 0; v < 8; ++v) {          // C: VGPR v -> M = v + 8*half
        int mrow = (rs * MR + r) * 16 + v + half * 8;
        float val = acc[r][q][v] + bv[q];
        if (relu) val = fmaxf(val, 0.f);
        C[(size_t)mrow * N + col] = val;
      }
    }
  }
}

// ---------------------------------------------------------------------------
// 7. pair scorer: one wave per 16 pairs.
//    feat = [src, dst, src*dst] (16x384 bf16 in LDS, 16B aligned rows)
//    hp   = relu(feat @ Wp1 + bp1)  -- 12x8 WMMAs, A via ds_load_b128,
//                                      B via pre-swizzled fragments
//    out  = hp . Wp2 + bp2          -- shfl_xor reduction
// ---------------------------------------------------------------------------
__global__ void pair_score_wmma(const float* __restrict__ z,      // [N,EMBED]
                                const int* __restrict__ pairs,    // [2,P]
                                const __bf16* __restrict__ Wp1swz,// swizzled [3E,E]
                                const float* __restrict__ bp1,    // [E]
                                const float* __restrict__ Wp2,    // [E]
                                const float* __restrict__ bp2,    // [1]
                                float* __restrict__ out) {
  __shared__ __attribute__((aligned(16))) __bf16 feat[16][3 * EMBED];
  int lane  = threadIdx.x;                  // 32 threads = one wave
  int ptile = __builtin_amdgcn_readfirstlane(blockIdx.x);

  for (int idx = lane; idx < 16 * 3 * EMBED; idx += 32) {
    int i = idx / (3 * EMBED);
    int c = idx % (3 * EMBED);
    int p = ptile * 16 + i;
    int s = pairs[p];
    int d = pairs[N_PAIRS + p];
    float v;
    if (c < EMBED)          v = z[(size_t)s * EMBED + c];
    else if (c < 2 * EMBED) v = z[(size_t)d * EMBED + (c - EMBED)];
    else                    v = z[(size_t)s * EMBED + (c - 2 * EMBED)] *
                                z[(size_t)d * EMBED + (c - 2 * EMBED)];
    feat[i][c] = (__bf16)v;
  }
  __syncthreads();

  int half = lane >> 4;
  int l15  = lane & 15;
  v8f acc[EMBED / 16] = {};                 // 8 column tiles
  for (int kk = 0; kk < 3 * EMBED; kk += 32) {
    int kt = kk >> 5;
    Frag a;
    const uint4* fp = (const uint4*)&feat[l15][kk + half * 8];
    a.q[0] = fp[0];
    a.q[1] = fp[2];
#pragma unroll
    for (int t = 0; t < EMBED / 16; ++t) {
      Frag b;
      const uint4* bp = (const uint4*)(Wp1swz + (((size_t)kt * (EMBED / 16) + t) * 32 + lane) * 16);
      b.q[0] = bp[0]; b.q[1] = bp[1];
      acc[t] = __builtin_amdgcn_wmma_f32_16x16x32_bf16(
          false, a.v, false, b.v, (short)0, acc[t], false, false);
    }
  }

  float part[8];
#pragma unroll
  for (int r = 0; r < 8; ++r) part[r] = 0.f;
#pragma unroll
  for (int t = 0; t < EMBED / 16; ++t) {
    int col = t * 16 + l15;
    float bb = bp1[col];
    float w2 = Wp2[col];
#pragma unroll
    for (int r = 0; r < 8; ++r) {
      float h = fmaxf(acc[t][r] + bb, 0.f);
      part[r] = fmaf(h, w2, part[r]);
    }
  }
#pragma unroll
  for (int m = 1; m < 16; m <<= 1)          // masks < 16: halves stay apart
#pragma unroll
    for (int r = 0; r < 8; ++r)
      part[r] += __shfl_xor(part[r], m, 32);

  if (l15 == 0) {
    float b2v = bp2[0];
#pragma unroll
    for (int r = 0; r < 8; ++r)
      out[ptile * 16 + half * 8 + r] = part[r] + b2v;
  }
}

// ---------------------------------------------------------------------------
// launch
// ---------------------------------------------------------------------------
extern "C" void kernel_launch(void* const* d_in, const int* in_sizes, int n_in,
                              void* d_out, int out_size, void* d_ws, size_t ws_size,
                              hipStream_t stream) {
  const int*   type_ids = (const int*)d_in[0];
  const int*   rows     = (const int*)d_in[1];
  const int*   cols     = (const int*)d_in[2];
  const float* vals     = (const float*)d_in[3];
  const int*   pairs    = (const int*)d_in[4];
  const float* node_emb = (const float*)d_in[5];
  const float* type_emb = (const float*)d_in[6];
  const float* W1  = (const float*)d_in[7];
  const float* b1  = (const float*)d_in[8];
  const float* W2  = (const float*)d_in[9];
  const float* b2  = (const float*)d_in[10];
  const float* Wp1 = (const float*)d_in[11];
  const float* bp1 = (const float*)d_in[12];
  const float* Wp2 = (const float*)d_in[13];
  const float* bp2 = (const float*)d_in[14];
  float* out = (float*)d_out;

  // 64B-aligned workspace sub-allocation
  uintptr_t p = (uintptr_t)d_ws;
  auto alloc = [&](size_t bytes) -> void* {
    p = (p + 63) & ~(uintptr_t)63;
    void* r = (void*)p;
    p += bytes;
    return r;
  };
  float*  bufA   = (float*) alloc((size_t)N_NODES * HIDDEN * 4);   // x, then h
  __bf16* bufBbf = (__bf16*)alloc((size_t)N_NODES * HIDDEN * 2);   // spmm out
  float*  bufZ   = (float*) alloc((size_t)N_NODES * EMBED * 4);    // z
  int*    rowptr = (int*)   alloc((size_t)(N_NODES + 1) * 4);
  __bf16* w1swz  = (__bf16*)alloc((size_t)HIDDEN * HIDDEN * 2);
  __bf16* w2swz  = (__bf16*)alloc((size_t)HIDDEN * EMBED * 2);
  __bf16* wp1swz = (__bf16*)alloc((size_t)3 * EMBED * EMBED * 2);

  // weight pre-swizzle into B-fragment order (tiny, once per launch)
  swizzle_w_kernel<<<(HIDDEN * HIDDEN + 255) / 256, 256, 0, stream>>>(W1, w1swz, HIDDEN, HIDDEN);
  swizzle_w_kernel<<<(HIDDEN * EMBED + 255) / 256, 256, 0, stream>>>(W2, w2swz, HIDDEN, EMBED);
  swizzle_w_kernel<<<(3 * EMBED * EMBED + 255) / 256, 256, 0, stream>>>(Wp1, wp1swz, 3 * EMBED, EMBED);

  // 1. x = node_emb + type_emb[ids]
  embed_kernel<<<(N_NODES * HIDDEN + 255) / 256, 256, 0, stream>>>(node_emb, type_emb, type_ids, bufA);
  // 2. row pointers from sorted rows
  rowptr_kernel<<<(N_NODES + 256) / 256, 256, 0, stream>>>(rows, rowptr);
  // 3. ax = spmm(x)  (bf16 out)
  spmm_kernel<<<N_NODES, HIDDEN / 4, 0, stream>>>(bufA, rowptr, cols, vals, bufBbf);
  // 4. h = relu(ax @ W1 + b1) -> bufA   (1250 blocks x 8 waves, exact)
  {
    int waves = (N_NODES / (16 * MR)) * (HIDDEN / 32);
    gemm_bias_wmma<<<waves / 8, 256, 0, stream>>>(bufBbf, w1swz, b1, bufA,
                                                  N_NODES, HIDDEN, HIDDEN, 1);
  }
  // 5. ah = spmm(h)  (bf16 out)
  spmm_kernel<<<N_NODES, HIDDEN / 4, 0, stream>>>(bufA, rowptr, cols, vals, bufBbf);
  // 6. z = ah @ W2 + b2 -> bufZ   (625 blocks x 8 waves, exact)
  {
    int waves = (N_NODES / (16 * MR)) * (EMBED / 32);
    gemm_bias_wmma<<<waves / 8, 256, 0, stream>>>(bufBbf, w2swz, b2, bufZ,
                                                  N_NODES, EMBED, HIDDEN, 0);
  }
  // 7. pair scoring
  pair_score_wmma<<<N_PAIRS / 16, 32, 0, stream>>>(bufZ, pairs, wp1swz, bp1, Wp2, bp2, out);
}
